// UniMPBlock_49727131353925
// MI455X (gfx1250) — matrix operand order repaired and checked by reading, not compiled, not run on hardware
//
#include <hip/hip_runtime.h>
#include <hip/hip_bf16.h>
#include <math.h>

typedef __bf16 bf16;
typedef __attribute__((ext_vector_type(16))) __bf16 v16bf;
typedef __attribute__((ext_vector_type(8)))  __bf16 v8bf;
typedef __attribute__((ext_vector_type(8)))  float   v8f;

// ---------------------------------------------------------------------------
// WMMA helpers (gfx1250, wave32): D(16x16,f32) = A(16x32,bf16) * B(32x16,bf16) + C
// ---------------------------------------------------------------------------

__device__ __forceinline__ v8f wmma_bf16(v16bf a, v16bf b, v8f c) {
    return __builtin_amdgcn_wmma_f32_16x16x32_bf16(
        /*neg_a=*/false, a, /*neg_b=*/false, b,
        /*c_mod=*/(short)0, c, /*reuse_a=*/false, /*reuse_b=*/false);
}

// A fragment: 16(M)x32(K) bf16 tile, row-major in LDS with leading dim ldm
// (ldm multiple of 8 so the 8-elem vectors are 16B aligned -> ds_load_b128).
// Lane L(0-15)  = row M=L: elems 0..7 = K 0..7,  elems 8..15 = K 16..23.
// Lane L+16     = row M=L: elems 0..7 = K 8..15, elems 8..15 = K 24..31.
__device__ __forceinline__ v16bf load_a_lds(const bf16* tile, int ldm, int kbase) {
    int lane = threadIdx.x & 31;
    const bf16* p = tile + (lane & 15) * ldm + kbase + (lane >> 4) * 8;
    v8bf lo = *(const v8bf*)p;
    v8bf hi = *(const v8bf*)(p + 16);
    v16bf a;
#pragma unroll
    for (int j = 0; j < 8; ++j) { a[j] = lo[j]; a[8 + j] = hi[j]; }
    return a;
}

// B fragment from the pre-swizzled bf16 weight buffer: each lane reads its 16
// bf16 values contiguously (32B -> 2x global_load_b128, fully coalesced).
__device__ __forceinline__ v16bf load_b_packed(const bf16* __restrict__ packed,
                                               int nct, int kc, int ct) {
    int lane = threadIdx.x & 31;
    const bf16* p = packed + ((size_t)(kc * nct + ct) * 32 + lane) * 16;
    return *(const v16bf*)p;
}

// Monotonic float<->uint mapping for atomicMax-based segment max.
__device__ __forceinline__ unsigned fkey_enc(float f) {
    unsigned u = __float_as_uint(f);
    return (u & 0x80000000u) ? ~u : (u | 0x80000000u);
}
__device__ __forceinline__ float fkey_dec(unsigned kk) {
    unsigned u = (kk & 0x80000000u) ? (kk & 0x7FFFFFFFu) : ~kk;
    return __uint_as_float(u);
}

__device__ __forceinline__ void atomic_fadd(float* p, float v) {
    __hip_atomic_fetch_add(p, v, __ATOMIC_RELAXED, __HIP_MEMORY_SCOPE_AGENT);
}

// ---------------------------------------------------------------------------
// P0: pack all 6 weight matrices into bf16 B-fragment layout.
// Matrices 0..4 (Wq,Wk,Wv,Wskip,We): 64x256  -> nkc=2, nct=16 (32 frags each)
// Matrix  5    (Wout):               256x64  -> nkc=8, nct=4  (32 frags)
// One wave per fragment; packed elem (kc,ct,lane,j) = W[kc*32+(lane>>4)*8+(j&7)+(j>>3)*16][ct*16+(lane&15)]
// ---------------------------------------------------------------------------
__global__ void __launch_bounds__(32) pack_kernel(
    const float* __restrict__ Wq, const float* __restrict__ Wk,
    const float* __restrict__ Wv, const float* __restrict__ Wsk,
    const float* __restrict__ We, const float* __restrict__ Wout,
    bf16* __restrict__ packed)
{
    const int f = blockIdx.x;
    const int lane = threadIdx.x;
    const float* W; int ldw, nct, kc, ct, m;
    if (f < 160) {
        m = f >> 5; int r = f & 31; kc = r >> 4; ct = r & 15; nct = 16; ldw = 256;
        W = (m == 0) ? Wq : (m == 1) ? Wk : (m == 2) ? Wv : (m == 3) ? Wsk : We;
    } else {
        m = 5; int r = f - 160; kc = r >> 2; ct = r & 3; nct = 4; ldw = 64;
        W = Wout;
    }
    bf16* dst = packed + (size_t)m * 16384 +
                ((size_t)(kc * nct + ct) * 32 + lane) * 16;
    const int c  = ct * 16 + (lane & 15);
    const int kb = kc * 32 + (lane >> 4) * 8;
#pragma unroll
    for (int j = 0; j < 16; ++j) dst[j] = (bf16)W[(size_t)(kb + (j & 7) + (j >> 3) * 16) * ldw + c];
}

// ---------------------------------------------------------------------------
// P1: node projections  q|k|v|skip = x @ W* + b*   (x:[N,64], W:[64,256])
// block = 128 threads (4 waves), one 16-row node tile per block.
// ---------------------------------------------------------------------------
__global__ void __launch_bounds__(128) proj_kernel(
    const float* __restrict__ x, const bf16* __restrict__ packed,
    const float* __restrict__ bq, const float* __restrict__ bk,
    const float* __restrict__ bv, const float* __restrict__ bsk,
    float* __restrict__ q, float* __restrict__ k,
    float* __restrict__ v, float* __restrict__ sk, int N)
{
    __shared__ bf16 xa[16 * 64];
    const int tid = threadIdx.x;
    const int n0 = blockIdx.x * 16;

    for (int i = tid; i < 16 * 64; i += 128) {
        int r = i >> 6, c = i & 63;
        xa[i] = (bf16)x[(size_t)(n0 + r) * 64 + c];
    }
    __syncthreads();

    const int w = tid >> 5, lane = tid & 31, col = lane & 15, g = lane >> 4;
    const v16bf a0 = load_a_lds(xa, 64, 0);
    const v16bf a1 = load_a_lds(xa, 64, 32);

#pragma unroll
    for (int m = 0; m < 4; ++m) {
        const float* b; float* o;
        if      (m == 0) { b = bq;  o = q;  }
        else if (m == 1) { b = bk;  o = k;  }
        else if (m == 2) { b = bv;  o = v;  }
        else             { b = bsk; o = sk; }
        const bf16* pw = packed + (size_t)m * 16384;
#pragma unroll
        for (int j = 0; j < 4; ++j) {
            int ct = w * 4 + j;
            int cb = ct * 16;
            v16bf b0 = load_b_packed(pw, 16, 0, ct);
            v16bf b1 = load_b_packed(pw, 16, 1, ct);
            v8f acc = {};
            acc = wmma_bf16(a0, b0, acc);
            acc = wmma_bf16(a1, b1, acc);
            float bias = b[cb + col];
#pragma unroll
            for (int i = 0; i < 8; ++i) {
                int row = g * 8 + i;
                o[(size_t)(n0 + row) * 256 + cb + col] = acc[i] + bias;
            }
        }
    }
}

// ---------------------------------------------------------------------------
// P2: per-edge  e = edge_attr @ We (WMMA, recomputed, never stored to HBM),
//     alpha[e,h] = 0.125 * dot(q[dst], k[src]+e); atomicMax into amax keys.
// block = 128 threads, 16 edges per block.
// ---------------------------------------------------------------------------
__global__ void __launch_bounds__(128) edge_alpha_kernel(
    const int* __restrict__ ei, const float* __restrict__ ea,
    const bf16* __restrict__ pWe,
    const float* __restrict__ q, const float* __restrict__ k,
    float* __restrict__ alpha, unsigned* __restrict__ amaxKey, int E)
{
    __shared__ bf16  eat[16 * 64];
    __shared__ float et[16 * 260];
    __shared__ float qt[16 * 260];
    __shared__ int   lsrc[16], ldst[16];

    const int tid = threadIdx.x;
    const size_t e0 = (size_t)blockIdx.x * 16;

    if (tid < 16) {
        lsrc[tid] = ei[e0 + tid];
        ldst[tid] = ei[(size_t)E + e0 + tid];
    }
    for (int i = tid; i < 16 * 64; i += 128) {
        int r = i >> 6, c = i & 63;
        eat[i] = (bf16)ea[(e0 + r) * 64 + c];
    }
    if ((int)blockIdx.x + 1 < (int)gridDim.x && tid == 0)
        __builtin_prefetch(ea + (e0 + 16) * 64, 0, 1);   // global_prefetch_b8
    __syncthreads();

    const int w = tid >> 5, lane = tid & 31, col = lane & 15, g = lane >> 4;
    const v16bf a0 = load_a_lds(eat, 64, 0);
    const v16bf a1 = load_a_lds(eat, 64, 32);
#pragma unroll
    for (int j = 0; j < 4; ++j) {
        int ct = w * 4 + j, cb = ct * 16;
        v16bf b0 = load_b_packed(pWe, 16, 0, ct);
        v16bf b1 = load_b_packed(pWe, 16, 1, ct);
        v8f acc = {};
        acc = wmma_bf16(a0, b0, acc);
        acc = wmma_bf16(a1, b1, acc);
#pragma unroll
        for (int i = 0; i < 8; ++i) et[(g * 8 + i) * 260 + cb + col] = acc[i];
    }
    __syncthreads();

    for (int p = tid; p < 16 * 256; p += 128) {           // e += k[src]; gather q[dst]
        int r = p >> 8, c = p & 255;
        et[r * 260 + c] += k[(size_t)lsrc[r] * 256 + c];
        qt[r * 260 + c]  = q[(size_t)ldst[r] * 256 + c];
    }
    __syncthreads();

    if (tid < 64) {                                       // (edge, head) dots (LDS only)
        int r = tid >> 2, h = tid & 3;
        const float* qrow = qt + r * 260 + h * 64;
        const float* erow = et + r * 260 + h * 64;
        float s = 0.f;
#pragma unroll 8
        for (int d = 0; d < 64; ++d) {
            int dd = (d + (h << 4)) & 63;                 // skew: break head bank conflicts
            s += qrow[dd] * erow[dd];
        }
        s *= 0.125f;                                      // 1/sqrt(64)
        alpha[(e0 + r) * 4 + h] = s;
        atomicMax(&amaxKey[(size_t)ldst[r] * 4 + h], fkey_enc(s));
    }
}

// ---------------------------------------------------------------------------
// P3: ex = exp(alpha - amax[dst]); atomicAdd into denom[N,H]; alpha <- ex
// ---------------------------------------------------------------------------
__global__ void softmax_ed_kernel(const int* __restrict__ ei,
                                  float* __restrict__ alpha,
                                  const unsigned* __restrict__ amaxKey,
                                  float* __restrict__ denom, int E)
{
    size_t i = (size_t)blockIdx.x * blockDim.x + threadIdx.x;
    if (i >= (size_t)E * 4) return;
    size_t e = i >> 2; int h = (int)(i & 3);
    int d = ei[(size_t)E + e];
    float amax = fkey_dec(amaxKey[(size_t)d * 4 + h]);
    if ((__float_as_uint(amax) & 0x7F800000u) == 0x7F800000u) amax = 0.f;  // non-finite -> 0
    float ex = __expf(alpha[i] - amax);
    alpha[i] = ex;
    atomic_fadd(&denom[(size_t)d * 4 + h], ex);
}

// ---------------------------------------------------------------------------
// P4: recompute e (WMMA), msg = (v[src]+e) * (ex / max(denom[dst],1e-16)),
//     atomic f32 scatter-add into agg[N,256] (L2-resident).
// ---------------------------------------------------------------------------
__global__ void __launch_bounds__(128) edge_agg_kernel(
    const int* __restrict__ ei, const float* __restrict__ ea,
    const bf16* __restrict__ pWe, const float* __restrict__ v,
    const float* __restrict__ exbuf, const float* __restrict__ denom,
    float* __restrict__ agg, int E)
{
    __shared__ bf16  eat[16 * 64];
    __shared__ float et[16 * 260];
    __shared__ int   lsrc[16], ldst[16];
    __shared__ float wgt[16 * 4];

    const int tid = threadIdx.x;
    const size_t e0 = (size_t)blockIdx.x * 16;

    if (tid < 16) {
        lsrc[tid] = ei[e0 + tid];
        ldst[tid] = ei[(size_t)E + e0 + tid];
    }
    for (int i = tid; i < 16 * 64; i += 128) {
        int r = i >> 6, c = i & 63;
        eat[i] = (bf16)ea[(e0 + r) * 64 + c];
    }
    if ((int)blockIdx.x + 1 < (int)gridDim.x && tid == 0)
        __builtin_prefetch(ea + (e0 + 16) * 64, 0, 1);
    __syncthreads();

    const int w = tid >> 5, lane = tid & 31, col = lane & 15, g = lane >> 4;
    const v16bf a0 = load_a_lds(eat, 64, 0);
    const v16bf a1 = load_a_lds(eat, 64, 32);
#pragma unroll
    for (int j = 0; j < 4; ++j) {
        int ct = w * 4 + j, cb = ct * 16;
        v16bf b0 = load_b_packed(pWe, 16, 0, ct);
        v16bf b1 = load_b_packed(pWe, 16, 1, ct);
        v8f acc = {};
        acc = wmma_bf16(a0, b0, acc);
        acc = wmma_bf16(a1, b1, acc);
#pragma unroll
        for (int i = 0; i < 8; ++i) et[(g * 8 + i) * 260 + cb + col] = acc[i];
    }
    __syncthreads();

    for (int p = tid; p < 16 * 256; p += 128) {           // e += v[src]
        int r = p >> 8, c = p & 255;
        et[r * 260 + c] += v[(size_t)lsrc[r] * 256 + c];
    }
    if (tid < 64) {
        int r = tid >> 2, h = tid & 3;
        float dn = denom[(size_t)ldst[r] * 4 + h];
        wgt[r * 4 + h] = exbuf[(e0 + r) * 4 + h] / fmaxf(dn, 1e-16f);
    }
    __syncthreads();

    for (int p = tid; p < 16 * 256; p += 128) {
        int r = p >> 8, c = p & 255;
        float val = et[r * 260 + c] * wgt[r * 4 + (c >> 6)];
        atomic_fadd(&agg[(size_t)ldst[r] * 256 + c], val);
    }
}

// ---------------------------------------------------------------------------
// P5: h = agg + skip; LayerNorm(256, biased var); out = elu(hn@Wout + bout + x)
// ---------------------------------------------------------------------------
__global__ void __launch_bounds__(128) out_kernel(
    const float* __restrict__ agg, const float* __restrict__ skipb,
    const float* __restrict__ ln_a, const float* __restrict__ ln_b,
    const bf16* __restrict__ pWout, const float* __restrict__ bout,
    const float* __restrict__ x, float* __restrict__ out, int N)
{
    __shared__ float ht[16 * 260];
    __shared__ bf16  hn[16 * 264];
    __shared__ float mean_s[16], rstd_s[16];

    const int tid = threadIdx.x;
    const int n0 = blockIdx.x * 16;

    for (int p = tid; p < 16 * 256; p += 128) {
        int r = p >> 8, c = p & 255;
        size_t gi = (size_t)(n0 + r) * 256 + c;
        ht[r * 260 + c] = agg[gi] + skipb[gi];
    }
    __syncthreads();

    if (tid < 16) {                                       // two-pass mean/var
        const float* row = ht + tid * 260;
        float s = 0.f;
        for (int c = 0; c < 256; ++c) s += row[c];
        float mean = s * (1.0f / 256.0f);
        float s2 = 0.f;
        for (int c = 0; c < 256; ++c) { float d = row[c] - mean; s2 += d * d; }
        mean_s[tid] = mean;
        rstd_s[tid] = rsqrtf(s2 * (1.0f / 256.0f) + 1e-5f);
    }
    __syncthreads();

    for (int p = tid; p < 16 * 256; p += 128) {
        int r = p >> 8, c = p & 255;
        float hv = ln_a[c] * (ht[r * 260 + c] - mean_s[r]) * rstd_s[r] + ln_b[c];
        hn[r * 264 + c] = (bf16)hv;
    }
    __syncthreads();

    const int w = tid >> 5, lane = tid & 31, col = lane & 15, g = lane >> 4;
    v8f acc = {};
#pragma unroll
    for (int kc = 0; kc < 8; ++kc) {
        v16bf a = load_a_lds(hn, 264, kc * 32);
        v16bf b = load_b_packed(pWout, 4, kc, w);
        acc = wmma_bf16(a, b, acc);
    }
#pragma unroll
    for (int i = 0; i < 8; ++i) {
        int row = g * 8 + i;
        int c = w * 16 + col;
        float val = acc[i] + bout[c] + x[(size_t)(n0 + row) * 64 + c];
        out[(size_t)(n0 + row) * 64 + c] = (val > 0.f) ? val : (__expf(val) - 1.0f);
    }
}

// ---------------------------------------------------------------------------
// Host-side launch
// ---------------------------------------------------------------------------
extern "C" void kernel_launch(void* const* d_in, const int* in_sizes, int n_in,
                              void* d_out, int out_size, void* d_ws, size_t ws_size,
                              hipStream_t stream) {
    (void)n_in; (void)out_size; (void)ws_size;
    const float* x    = (const float*)d_in[0];
    const int*   ei   = (const int*)  d_in[1];
    const float* ea   = (const float*)d_in[2];
    const float* Wq   = (const float*)d_in[3];
    const float* bq   = (const float*)d_in[4];
    const float* Wk   = (const float*)d_in[5];
    const float* bk   = (const float*)d_in[6];
    const float* Wv   = (const float*)d_in[7];
    const float* bv   = (const float*)d_in[8];
    const float* We   = (const float*)d_in[9];
    const float* Wsk  = (const float*)d_in[10];
    const float* bsk  = (const float*)d_in[11];
    const float* ln_a = (const float*)d_in[12];
    const float* ln_b = (const float*)d_in[13];
    const float* Wout = (const float*)d_in[14];
    const float* bout = (const float*)d_in[15];
    float* out = (float*)d_out;

    const int N = in_sizes[0] / 64;
    const int E = in_sizes[2] / 64;

    // Workspace layout (fp32 words; every region size is a multiple of 8
    // floats so the bf16 pack buffer stays 32B aligned):
    float* ws = (float*)d_ws;
    size_t off = 0;
    float*    q       = ws + off; off += (size_t)N * 256;
    float*    k       = ws + off; off += (size_t)N * 256;
    float*    v       = ws + off; off += (size_t)N * 256;
    float*    sk      = ws + off; off += (size_t)N * 256;
    float*    alpha   = ws + off; off += (size_t)E * 4;
    unsigned* amaxKey = (unsigned*)(ws + off); off += (size_t)N * 4;
    float*    denom   = ws + off; off += (size_t)N * 4;
    float*    agg     = ws + off; off += (size_t)N * 256;
    bf16*     packedW = (bf16*)(ws + off);     // 6 * 16384 bf16 = 192 KiB

    hipMemsetAsync(amaxKey, 0, (size_t)N * 4 * sizeof(unsigned), stream); // key 0 lies below enc(-inf)
    hipMemsetAsync(denom,   0, (size_t)N * 4 * sizeof(float),    stream);
    hipMemsetAsync(agg,     0, (size_t)N * 256 * sizeof(float),  stream);

    const int nb  = N / 16;   // N=50000 -> 3125 tiles
    const int ebl = E / 16;   // E=800000 -> 50000 tiles

    pack_kernel<<<192, 32, 0, stream>>>(Wq, Wk, Wv, Wsk, We, Wout, packedW);
    proj_kernel<<<nb, 128, 0, stream>>>(x, packedW, bq, bk, bv, bsk, q, k, v, sk, N);
    edge_alpha_kernel<<<ebl, 128, 0, stream>>>(ei, ea, packedW + (size_t)4 * 16384,
                                               q, k, alpha, amaxKey, E);
    size_t tot = (size_t)E * 4;
    softmax_ed_kernel<<<(int)((tot + 255) / 256), 256, 0, stream>>>(ei, alpha, amaxKey,
                                                                    denom, E);
    edge_agg_kernel<<<ebl, 128, 0, stream>>>(ei, ea, packedW + (size_t)4 * 16384,
                                             v, alpha, denom, agg, E);
    out_kernel<<<nb, 128, 0, stream>>>(agg, sk, ln_a, ln_b,
                                       packedW + (size_t)5 * 16384, bout, x, out, N);
}
